// BidirectionalCrossAttention_78898549227847
// MI455X (gfx1250) — matrix operand order repaired and checked by reading, not compiled, not run on hardware
//
#include <hip/hip_runtime.h>
#include <hip/hip_bf16.h>

// ---------------- problem constants ----------------
#define BB   4
#define CC   256
#define KH   8
#define DD   32
#define NPIX 16384          // H*W = 128*128
#define TP   64             // pixels per tile
#define NT   (NPIX / TP)    // 256 tiles

// LDS row strides (in halves)
#define LDT 264             // ref/x tile transposed [TP][CC], padded
#define LDE 72              // e/v head tiles [32][TP], padded
#define LDO 520             // concat out tile [TP][2C], padded

// ---------------- workspace layout ----------------
// bf16 weights (ushort elements)
#define WQ_OFF   0
#define WK1_OFF  65536
#define WK2_OFF  131072
#define WV1_OFF  196608
#define WV2_OFF  262144
#define WO_OFF   327680
#define WBF_HALVES 458752
// byte offsets after weights
#define RMAX_BYTE  (WBF_HALVES * 2)           // unsigned[2048]  = [B][2][256]
#define RSUM_BYTE  (RMAX_BYTE + 2048 * 4)     // float[2048]
#define CTX_BYTE   (RSUM_BYTE + 2048 * 4)     // float[65536]    = [B][2][8][32][32]
#define CTXBF_BYTE (CTX_BYTE + 65536 * 4)     // ushort[65536]

typedef __attribute__((ext_vector_type(16))) __bf16 v16bf;
typedef __attribute__((ext_vector_type(8)))  float  v8f;

union ABFrag { v16bf v; unsigned u[8]; };
union Acc    { v8f v; float f[8]; };

__device__ __forceinline__ unsigned short f32_to_bf16(float f) {
  union { float f; unsigned u; } a; a.f = f;
  unsigned u = a.u;
  u += 0x7FFFu + ((u >> 16) & 1u);          // round-to-nearest-even
  return (unsigned short)(u >> 16);
}
// monotone float<->uint keys for atomicMax over signed floats
__device__ __forceinline__ unsigned enc_f32(float f) {
  union { float f; unsigned u; } a; a.f = f;
  unsigned u = a.u;
  return u ^ ((u & 0x80000000u) ? 0xFFFFFFFFu : 0x80000000u);
}
__device__ __forceinline__ float dec_f32(unsigned e) {
  unsigned u = e ^ ((e & 0x80000000u) ? 0x80000000u : 0xFFFFFFFFu);
  union { unsigned u; float f; } a; a.u = u;
  return a.f;
}

// A fragment: source S[M][K] row-major (K contiguous), bf16.
// lanes 0-15: M=m0+lane, elems 0..7 -> K=k0..k0+7, elems 8..15 -> K=k0+16..k0+23
// lanes 16-31: same M, K ranges +8
__device__ __forceinline__ v16bf load_a_frag(const unsigned short* S, int ld, int m0, int k0) {
  unsigned lane = threadIdx.x & 31u;
  int m = m0 + (int)(lane & 15u);
  int kb = k0 + (int)((lane >> 4) << 3);
  ABFrag r;
  const unsigned short* row = S + (size_t)m * ld;
#pragma unroll
  for (int j = 0; j < 4; ++j) {
    r.u[j]     = *(const unsigned*)(row + kb + 2 * j);
    r.u[4 + j] = *(const unsigned*)(row + kb + 16 + 2 * j);
  }
  return r.v;
}
// B fragment: source T[N][K] row-major (K contiguous), bf16.
// lanes 0-15: N=n0+lane, elems 0..7 -> K=k0..k0+7, elems 8..15 -> K=k0+16..k0+23 ; lanes 16-31: same N, K ranges +8
__device__ __forceinline__ v16bf load_b_frag(const unsigned short* T, int ld, int k0, int n0) {
  unsigned lane = threadIdx.x & 31u;
  int n = n0 + (int)(lane & 15u);
  int kb = k0 + (int)((lane >> 4) << 3);
  ABFrag r;
  const unsigned short* row = T + (size_t)n * ld + kb;
#pragma unroll
  for (int j = 0; j < 4; ++j) {
    r.u[j]     = *(const unsigned*)(row + 2 * j);
    r.u[4 + j] = *(const unsigned*)(row + 16 + 2 * j);
  }
  return r.v;
}

__device__ __forceinline__ v8f wmma_bf16(v16bf a, v16bf b, v8f c) {
  return __builtin_amdgcn_wmma_f32_16x16x32_bf16(false, a, false, b, (short)0, c, false, false);
}

// ---------------- kernel 1: weight conversion + accumulator init ----------------
__global__ void __launch_bounds__(256)
prep_kernel(const float* Wq, const float* Wk1, const float* Wk2,
            const float* Wv1, const float* Wv2, const float* Wo,
            unsigned short* wbf, unsigned* rmax, float* rsum, float* gctx) {
  int i = blockIdx.x * blockDim.x + threadIdx.x;
  if (i < 65536) {
    wbf[WQ_OFF  + i] = f32_to_bf16(Wq[i]);
    wbf[WK1_OFF + i] = f32_to_bf16(Wk1[i]);
    wbf[WK2_OFF + i] = f32_to_bf16(Wk2[i]);
    wbf[WV1_OFF + i] = f32_to_bf16(Wv1[i]);
    wbf[WV2_OFF + i] = f32_to_bf16(Wv2[i]);
    gctx[i] = 0.0f;
  }
  if (i < 131072) wbf[WO_OFF + i] = f32_to_bf16(Wo[i]);
  if (i < 2048) { rmax[i] = 0u; rsum[i] = 0.0f; }
}

// ---------------- kernel 2: per-channel max of k over n (softmax-over-n stats) ----------------
__global__ void __launch_bounds__(256)
pass_rowmax(const float* ref1, const float* ref2, const unsigned short* wbf, unsigned* rmax) {
  extern __shared__ unsigned short sm_rt[]; // [TP][LDT]
  const int b = blockIdx.y, br = blockIdx.z;
  const int n0 = blockIdx.x * TP;
  const float* ref = br ? ref2 : ref1;
  const unsigned short* wk = wbf + (br ? WK2_OFF : WK1_OFF);

  for (int idx = threadIdx.x; idx < CC * TP; idx += 256) {
    int c = idx >> 6, p = idx & (TP - 1);
    const float* src = ref + (size_t)(b * CC + c) * NPIX + n0 + p;
    __builtin_prefetch(src + TP, 0, 1);
    sm_rt[p * LDT + c] = f32_to_bf16(*src);
  }
  __syncthreads();

  const int w = threadIdx.x >> 5;
  const unsigned lane = threadIdx.x & 31u, hi = lane >> 4, ln = lane & 15u;
  unsigned* dst = rmax + (size_t)(b * 2 + br) * CC;

#pragma unroll 1
  for (int i = 0; i < 8; ++i) {
    int t = w * 8 + i;
    int mt = t >> 2, nt = t & 3;
    Acc acc = {};
#pragma unroll
    for (int kk = 0; kk < 8; ++kk) {
      v16bf a = load_a_frag(wk, CC, mt * 16, kk * 32);
      v16bf bb = load_b_frag(sm_rt, LDT, kk * 32, nt * 16);
      acc.v = wmma_bf16(a, bb, acc.v);
    }
    float m[8];
#pragma unroll
    for (int r = 0; r < 8; ++r) m[r] = acc.f[r];
#pragma unroll
    for (int mask = 1; mask <= 8; mask <<= 1)
#pragma unroll
      for (int r = 0; r < 8; ++r) m[r] = fmaxf(m[r], __shfl_xor(m[r], mask, 32));
    if (ln == 0) {
#pragma unroll
      for (int r = 0; r < 8; ++r) {
        int c = mt * 16 + (int)hi * 8 + r;
        atomicMax(&dst[c], enc_f32(m[r]));
      }
    }
  }
}

// ---------------- kernel 3: ctx += softmax-num(k) * v^T ; rowsum += sum(e) ----------------
__global__ void __launch_bounds__(256)
pass_ctx(const float* ref1, const float* ref2, const unsigned short* wbf,
         const unsigned* rmaxg, float* rsumg, float* gctx) {
  extern __shared__ unsigned short sm[];
  unsigned short* rt = sm;                       // [TP][LDT]
  unsigned short* eL = sm + TP * LDT;            // [32][LDE]
  unsigned short* vL = eL + 32 * LDE;            // [32][LDE]

  const int b = blockIdx.y, br = blockIdx.z;
  const int n0 = blockIdx.x * TP;
  const float* ref = br ? ref2 : ref1;
  const unsigned short* wk = wbf + (br ? WK2_OFF : WK1_OFF);
  const unsigned short* wv = wbf + (br ? WV2_OFF : WV1_OFF);
  const unsigned* rmax = rmaxg + (size_t)(b * 2 + br) * CC;
  float* rsum = rsumg + (size_t)(b * 2 + br) * CC;

  for (int idx = threadIdx.x; idx < CC * TP; idx += 256) {
    int c = idx >> 6, p = idx & (TP - 1);
    const float* src = ref + (size_t)(b * CC + c) * NPIX + n0 + p;
    __builtin_prefetch(src + TP, 0, 1);
    rt[p * LDT + c] = f32_to_bf16(*src);
  }
  __syncthreads();

  const int w = threadIdx.x >> 5;
  const unsigned lane = threadIdx.x & 31u, hi = lane >> 4, ln = lane & 15u;
  const int mt = w >> 2, nt = w & 3;             // each wave: one 16x16 tile of [32 x TP]
  const int p = nt * 16 + (int)ln;

#pragma unroll 1
  for (int h = 0; h < KH; ++h) {
    // ---- k projection for this head ----
    Acc ka = {};
#pragma unroll
    for (int kk = 0; kk < 8; ++kk) {
      v16bf a = load_a_frag(wk, CC, h * 32 + mt * 16, kk * 32);
      v16bf bb = load_b_frag(rt, LDT, kk * 32, nt * 16);
      ka.v = wmma_bf16(a, bb, ka.v);
    }
    // ---- e = exp(k - rowmax) ; stage to LDS ; rowsum atomics ----
    float e[8];
#pragma unroll
    for (int r = 0; r < 8; ++r) {
      int c = h * 32 + mt * 16 + (int)hi * 8 + r;
      float mx = dec_f32(rmax[c]);
      e[r] = __expf(ka.f[r] - mx);
      eL[(mt * 16 + (int)hi * 8 + r) * LDE + p] = f32_to_bf16(e[r]);
    }
    float s[8];
#pragma unroll
    for (int r = 0; r < 8; ++r) s[r] = e[r];
#pragma unroll
    for (int mask = 1; mask <= 8; mask <<= 1)
#pragma unroll
      for (int r = 0; r < 8; ++r) s[r] += __shfl_xor(s[r], mask, 32);
    if (ln == 0) {
#pragma unroll
      for (int r = 0; r < 8; ++r)
        atomicAdd(&rsum[h * 32 + mt * 16 + (int)hi * 8 + r], s[r]);
    }
    // ---- v projection for this head ----
    Acc va = {};
#pragma unroll
    for (int kk = 0; kk < 8; ++kk) {
      v16bf a = load_a_frag(wv, CC, h * 32 + mt * 16, kk * 32);
      v16bf bb = load_b_frag(rt, LDT, kk * 32, nt * 16);
      va.v = wmma_bf16(a, bb, va.v);
    }
#pragma unroll
    for (int r = 0; r < 8; ++r)
      vL[(mt * 16 + (int)hi * 8 + r) * LDE + p] = f32_to_bf16(va.f[r]);
    __syncthreads();

    // ---- ctx[c,d] += sum_p e[c,p] * v[d,p]  (M=c, N=d, K=p) ----
    if (w < 4) {
      int cmt = w >> 1, dnt = w & 1;
      Acc cx = {};
#pragma unroll
      for (int kk = 0; kk < TP / 32; ++kk) {
        v16bf a = load_a_frag(eL, LDE, cmt * 16, kk * 32);
        v16bf bb = load_b_frag(vL, LDE, kk * 32, dnt * 16);
        cx.v = wmma_bf16(a, bb, cx.v);
      }
      float* dst = gctx + (size_t)(((b * 2 + br) * KH + h) * DD) * DD;
#pragma unroll
      for (int r = 0; r < 8; ++r) {
        int cl = cmt * 16 + (int)hi * 8 + r;
        int d = dnt * 16 + (int)ln;
        atomicAdd(&dst[cl * DD + d], cx.f[r]);
      }
    }
    __syncthreads();
  }
}

// ---------------- kernel 4: ctx_bf16 = ctx / rowsum ----------------
__global__ void __launch_bounds__(256)
pass_norm(const float* gctx, const float* rsum, unsigned short* ctxbf) {
  int i = blockIdx.x * blockDim.x + threadIdx.x;
  if (i < 65536) {
    int c = (i >> 5) & 31;
    int grp = i >> 10;                          // (b*2+br)*8 + h
    float s = rsum[grp * 32 + c];
    ctxbf[i] = f32_to_bf16(gctx[i] / s);
  }
}

// ---------------- kernel 5: q softmax, out_h = ctx*q^T, fused Wo projection ----------------
__global__ void __launch_bounds__(256)
pass_out(const float* x, const unsigned short* wbf, const unsigned short* ctxbf, float* out) {
  extern __shared__ unsigned short sm[];
  unsigned short* bufA = sm;                     // stage1-2: xT [TP][LDT]; stage3-4: outT [TP][LDO]
  unsigned short* qal  = sm + TP * LDO;          // q (softmaxed) [TP][LDT]
  unsigned short* ctxl = qal + TP * LDT;         // bf16 ctx for this batch: [2][8][32][32]

  const int b = blockIdx.y;
  const int n0 = blockIdx.x * TP;

  for (int idx = threadIdx.x; idx < CC * TP; idx += 256) {
    int c = idx >> 6, p = idx & (TP - 1);
    const float* src = x + (size_t)(b * CC + c) * NPIX + n0 + p;
    __builtin_prefetch(src + TP, 0, 1);
    bufA[p * LDT + c] = f32_to_bf16(*src);
  }
  for (int idx = threadIdx.x; idx < 16384; idx += 256)
    ctxl[idx] = ctxbf[(size_t)b * 16384 + idx];
  __syncthreads();

  const int w = threadIdx.x >> 5;                // wave w owns head w
  const unsigned lane = threadIdx.x & 31u, hi = lane >> 4, ln = lane & 15u;

  // ---- stage 2: q = Wq @ x, then softmax over head dim (32 rows) ----
  const unsigned short* wq = wbf + WQ_OFF;
  Acc qa[2][4] = {};
#pragma unroll 1
  for (int kk = 0; kk < 8; ++kk) {
#pragma unroll
    for (int mt = 0; mt < 2; ++mt) {
      v16bf a = load_a_frag(wq, CC, w * 32 + mt * 16, kk * 32);
#pragma unroll
      for (int nt = 0; nt < 4; ++nt) {
        v16bf bb = load_b_frag(bufA, LDT, kk * 32, nt * 16);
        qa[mt][nt].v = wmma_bf16(a, bb, qa[mt][nt].v);
      }
    }
  }
#pragma unroll
  for (int nt = 0; nt < 4; ++nt) {
    float mx = -3.4e38f;
#pragma unroll
    for (int mt = 0; mt < 2; ++mt)
#pragma unroll
      for (int r = 0; r < 8; ++r) mx = fmaxf(mx, qa[mt][nt].f[r]);
    mx = fmaxf(mx, __shfl_xor(mx, 16, 32));
    float s = 0.0f, e[2][8];
#pragma unroll
    for (int mt = 0; mt < 2; ++mt)
#pragma unroll
      for (int r = 0; r < 8; ++r) { e[mt][r] = __expf(qa[mt][nt].f[r] - mx); s += e[mt][r]; }
    s += __shfl_xor(s, 16, 32);
    float inv = 1.0f / s;
    int p = nt * 16 + (int)ln;
#pragma unroll
    for (int mt = 0; mt < 2; ++mt)
#pragma unroll
      for (int r = 0; r < 8; ++r)
        qal[p * LDT + w * 32 + mt * 16 + (int)hi * 8 + r] = f32_to_bf16(e[mt][r] * inv);
  }
  __syncthreads();

  // ---- stage 3: out_h[c,p] = sum_d ctx[c,d] * q[p,d]  (single K=32 WMMA per tile) ----
#pragma unroll 1
  for (int i = 0; i < 16; ++i) {
    int t = w * 16 + i;
    int br = t >> 6, rem = t & 63;
    int h = rem >> 3, mt = (rem >> 2) & 1, nt = rem & 3;
    const unsigned short* actx = ctxl + (br * KH + h) * 1024;   // [32][32]
    Acc oa = {};
    v16bf a = load_a_frag(actx, DD, mt * 16, 0);
    v16bf bb = load_b_frag(qal + h * 32, LDT, 0, nt * 16);
    oa.v = wmma_bf16(a, bb, oa.v);
    int p = nt * 16 + (int)ln;
#pragma unroll
    for (int r = 0; r < 8; ++r)
      bufA[p * LDO + br * 256 + h * 32 + mt * 16 + (int)hi * 8 + r] = f32_to_bf16(oa.f[r]);
  }
  __syncthreads();

  // ---- stage 4: out = Wo[256,512] @ concat(out1,out2)[512,p] ----
  const unsigned short* wo = wbf + WO_OFF;
#pragma unroll 1
  for (int i = 0; i < 8; ++i) {
    int t = w * 8 + i;
    int mt = t >> 2, nt = t & 3;
    Acc oc = {};
#pragma unroll
    for (int kk = 0; kk < 16; ++kk) {
      v16bf a = load_a_frag(wo, 2 * CC, mt * 16, kk * 32);
      v16bf bb = load_b_frag(bufA, LDO, kk * 32, nt * 16);
      oc.v = wmma_bf16(a, bb, oc.v);
    }
    int p = nt * 16 + (int)ln;
#pragma unroll
    for (int r = 0; r < 8; ++r) {
      int c = mt * 16 + (int)hi * 8 + r;
      out[(size_t)(b * CC + c) * NPIX + n0 + p] = oc.f[r];
    }
  }
}

// ---------------- host launcher ----------------
extern "C" void kernel_launch(void* const* d_in, const int* in_sizes, int n_in,
                              void* d_out, int out_size, void* d_ws, size_t ws_size,
                              hipStream_t stream) {
  const float* x    = (const float*)d_in[0];
  const float* ref1 = (const float*)d_in[1];
  const float* ref2 = (const float*)d_in[2];
  const float* Wq   = (const float*)d_in[3];
  const float* Wk1  = (const float*)d_in[4];
  const float* Wk2  = (const float*)d_in[5];
  const float* Wv1  = (const float*)d_in[6];
  const float* Wv2  = (const float*)d_in[7];
  const float* Wo   = (const float*)d_in[8];
  float* out = (float*)d_out;

  unsigned short* wbf  = (unsigned short*)d_ws;
  unsigned* rmax       = (unsigned*)((char*)d_ws + RMAX_BYTE);
  float* rsum          = (float*)((char*)d_ws + RSUM_BYTE);
  float* gctx          = (float*)((char*)d_ws + CTX_BYTE);
  unsigned short* ctxbf = (unsigned short*)((char*)d_ws + CTXBF_BYTE);

  const size_t smem_rmax = (size_t)TP * LDT * 2;                         // 33,792 B
  const size_t smem_ctx  = (size_t)(TP * LDT + 2 * 32 * LDE) * 2;        // 43,008 B
  const size_t smem_out  = (size_t)(TP * LDO + TP * LDT + 16384) * 2;    // 133,120 B

  (void)hipFuncSetAttribute((const void*)pass_rowmax,
      hipFuncAttributeMaxDynamicSharedMemorySize, (int)smem_rmax);
  (void)hipFuncSetAttribute((const void*)pass_ctx,
      hipFuncAttributeMaxDynamicSharedMemorySize, (int)smem_ctx);
  (void)hipFuncSetAttribute((const void*)pass_out,
      hipFuncAttributeMaxDynamicSharedMemorySize, (int)smem_out);

  prep_kernel<<<512, 256, 0, stream>>>(Wq, Wk1, Wk2, Wv1, Wv2, Wo, wbf, rmax, rsum, gctx);

  dim3 grid2(NT, BB, 2);
  pass_rowmax<<<grid2, 256, smem_rmax, stream>>>(ref1, ref2, wbf, rmax);
  pass_ctx<<<grid2, 256, smem_ctx, stream>>>(ref1, ref2, wbf, rmax, rsum, gctx);

  pass_norm<<<256, 256, 0, stream>>>(gctx, rsum, ctxbf);

  dim3 grid5(NT, BB, 1);
  pass_out<<<grid5, 256, smem_out, stream>>>(x, wbf, ctxbf, out);
}
